// StackedPointAttention_6665789243940
// MI455X (gfx1250) — matrix operand order repaired and checked by reading, not compiled, not run on hardware
//
#include <hip/hip_runtime.h>
#include <hip/hip_bf16.h>

// ---------------------------------------------------------------------------
// Problem constants (match reference)
// ---------------------------------------------------------------------------
#define BATCH 4
#define C_IN  128
#define C_OUT 256
#define NPTS  4096
#define DQK   64
#define NLAYR 4

typedef __bf16 v16bf __attribute__((ext_vector_type(16)));
typedef float  v8f   __attribute__((ext_vector_type(8)));
typedef unsigned int u32x4 __attribute__((ext_vector_type(4)));

union FragBF { v16bf v; u32x4 q[2]; unsigned int w[8]; unsigned short h[16]; };
union FragF  { v8f   v; float f[8]; };

__device__ __forceinline__ unsigned short f2bf(float f) {
  unsigned int x = __float_as_uint(f);
  x += 0x7FFFu + ((x >> 16) & 1u);          // round-to-nearest-even
  return (unsigned short)(x >> 16);
}

// A-fragment (16x32 bf16, row-major source): lane owns row m; needs K pairs
// {kA..kA+7} and {kA+16..kA+23} where kA = k0 + 8*(lane>=16).  Both runs are
// contiguous -> two b128 loads.
__device__ __forceinline__ void loadA(FragBF& f, const unsigned short* row, int k) {
  const u32x4* p = (const u32x4*)(row + k);
  f.q[0] = p[0];          // K = k .. k+7
  f.q[1] = p[2];          // K = k+16 .. k+23
}
// B-fragment (32x16 bf16) from B^T row-major: lane owns column n; needs K
// {kB..kB+7} and {kB+16..kB+23} where kB = k0 + 8*(lane>=16) -> two b128 loads.
__device__ __forceinline__ void loadB(FragBF& f, const unsigned short* p) {
  const u32x4* q = (const u32x4*)p;
  f.q[0] = q[0];
  f.q[1] = q[2];
}

// ---------------------------------------------------------------------------
// bf16 WMMA GEMM:  C[M,N] = A[M,K] (row-major) * B[K,N], B given as
// BT[N,K] row-major.  1 wave -> 32x64 tile (2 A-frags x 4 B-frags = 8 WMMAs
// per 32-wide K step), 4 waves/block -> 128x64 block tile.
// MODE 0: C = acc     MODE 1: C = acc + bias[m]     MODE 2: C = resid - acc
// ---------------------------------------------------------------------------
template <int MODE>
__global__ void __launch_bounds__(128)
gemm_bf16_wmma(const unsigned short* __restrict__ A,
               const unsigned short* __restrict__ BT,
               float* __restrict__ C,
               const float* __restrict__ bias,
               const float* __restrict__ resid,
               int M, int N, int Kd,
               int lda, int ldbt, int ldc, int ldr,
               long long sA, long long sBT, long long sC, long long sR)
{
  const int lane = threadIdx.x & 31;
  const int wv   = threadIdx.x >> 5;
  const int nT0  = blockIdx.x * 64;
  const int mT0  = (blockIdx.y * 4 + wv) * 32;
  const long long bz = blockIdx.z;
  A  += bz * sA;
  BT += bz * sBT;
  C  += bz * sC;
  if (MODE == 2) resid += bz * sR;

  const int half = lane >> 4;                 // 0 or 1
  const int l16  = lane & 15;
  const int kA   = half * 8;                  // A k-offset inside 32-step
  const int kB   = half * 8;                  // B k-offset inside 32-step

  const unsigned short* Arow0 = A + (size_t)(mT0 + l16) * lda;
  const unsigned short* Arow1 = A + (size_t)(mT0 + 16 + l16) * lda;

  FragF acc[2][4];
#pragma unroll
  for (int mi = 0; mi < 2; ++mi)
#pragma unroll
    for (int t = 0; t < 4; ++t)
#pragma unroll
      for (int r = 0; r < 8; ++r) acc[mi][t].f[r] = 0.0f;

  for (int k0 = 0; k0 < Kd; k0 += 32) {
    FragBF a0, a1;
    loadA(a0, Arow0, k0 + kA);
    loadA(a1, Arow1, k0 + kA);
#pragma unroll
    for (int t = 0; t < 4; ++t) {
      FragBF b;
      loadB(b, BT + (size_t)(nT0 + t * 16 + l16) * ldbt + k0 + kB);
      acc[0][t].v = __builtin_amdgcn_wmma_f32_16x16x32_bf16(
          false, a0.v, false, b.v, (short)0, acc[0][t].v, false, false);
      acc[1][t].v = __builtin_amdgcn_wmma_f32_16x16x32_bf16(
          false, a1.v, false, b.v, (short)0, acc[1][t].v, false, false);
    }
  }

#pragma unroll
  for (int mi = 0; mi < 2; ++mi) {
#pragma unroll
    for (int r = 0; r < 8; ++r) {
      const int m = mT0 + mi * 16 + r + half * 8;
      float badd = 0.0f;
      if (MODE == 1) badd = bias[m];
#pragma unroll
      for (int t = 0; t < 4; ++t) {
        const int n = nT0 + t * 16 + l16;
        float v = acc[mi][t].f[r];
        if (MODE == 1) v += badd;
        if (MODE == 2) v = resid[(size_t)m * ldr + n] - v;
        C[(size_t)m * ldc + n] = v;
      }
    }
  }
}

// ---------------------------------------------------------------------------
// LDS-tiled transpose + f32->bf16 convert with optional per-column L1 scale:
//   out[c, r] = f2bf( in[r, c] * (colscale ? 1/(1e-9 + colscale[c]) : 1) )
// in : [R, Cc] f32 row-major;  out : [Cc, R] bf16 row-major.
// Coalesced reads AND writes; 33-stride LDS tile avoids bank conflicts.
// ---------------------------------------------------------------------------
__global__ void __launch_bounds__(256)
transpose_scale_bf16(const float* __restrict__ in,
                     unsigned short* __restrict__ out,
                     int R, int Cc,
                     long long sIn, long long sOut,
                     const float* __restrict__ colscale)
{
  __shared__ float tile[32][33];
  in  += (long long)blockIdx.z * sIn;
  out += (long long)blockIdx.z * sOut;
  const int c0 = blockIdx.x * 32;
  const int r0 = blockIdx.y * 32;
  const int tx = threadIdx.x;
#pragma unroll
  for (int i = threadIdx.y; i < 32; i += 8)
    tile[i][tx] = in[(size_t)(r0 + i) * Cc + c0 + tx];
  __syncthreads();
#pragma unroll
  for (int i = threadIdx.y; i < 32; i += 8) {
    const int c = c0 + i;
    const float s = colscale ? (1.0f / (1e-9f + colscale[c])) : 1.0f;
    out[(size_t)c * R + r0 + tx] = f2bf(tile[tx][i] * s);
  }
}

// ---------------------------------------------------------------------------
// BatchNorm (training-mode) stats over (B, N) per channel
// ---------------------------------------------------------------------------
__global__ void __launch_bounds__(256)
bn_stats_kernel(const float* __restrict__ x, float* __restrict__ mean,
                float* __restrict__ rstd, int C, int N, int Bn, float eps)
{
  const int c = blockIdx.x;
  float s = 0.f, s2 = 0.f;
  for (int b = 0; b < Bn; ++b) {
    const float* p = x + ((size_t)b * C + c) * N;
    for (int n = threadIdx.x; n < N; n += blockDim.x) {
      float v = p[n];
      s += v; s2 += v * v;
    }
  }
  __shared__ float sh[256], sh2[256];
  sh[threadIdx.x] = s; sh2[threadIdx.x] = s2;
  __syncthreads();
  for (int st = 128; st > 0; st >>= 1) {
    if ((int)threadIdx.x < st) {
      sh[threadIdx.x]  += sh[threadIdx.x + st];
      sh2[threadIdx.x] += sh2[threadIdx.x + st];
    }
    __syncthreads();
  }
  if (threadIdx.x == 0) {
    const float inv = 1.0f / ((float)Bn * (float)N);
    const float mu  = sh[0] * inv;
    const float var = sh2[0] * inv - mu * mu;
    mean[c] = mu;
    rstd[c] = rsqrtf(var + eps);
  }
}

// BN apply + ReLU; writes f32 result and optionally a slice of the
// concatenated output.  (bf16 transposed copy is produced separately.)
__global__ void __launch_bounds__(256)
bn_apply_relu(const float* __restrict__ x, const float* __restrict__ mean,
              const float* __restrict__ rstd, const float* __restrict__ g,
              const float* __restrict__ bta,
              float* __restrict__ outf,
              float* __restrict__ out2, long long out2_bstride,
              int C, int N, int total)
{
  const int i = blockIdx.x * blockDim.x + threadIdx.x;
  if (i >= total) return;
  const int n = i % N;
  const int c = (i / N) % C;
  const int b = i / (N * C);
  float v = (x[i] - mean[c]) * rstd[c] * g[c] + bta[c];
  v = fmaxf(v, 0.0f);
  outf[i] = v;
  if (out2) out2[(long long)b * out2_bstride + (size_t)c * N + n] = v;
}

__global__ void __launch_bounds__(256)
convert_bf16_kernel(const float* __restrict__ in,
                    unsigned short* __restrict__ out, int total)
{
  const int i = blockIdx.x * blockDim.x + threadIdx.x;
  if (i < total) out[i] = f2bf(in[i]);
}

// Row softmax over a 4096-wide row (one block per row).
__global__ void __launch_bounds__(256)
softmax_row_kernel(float* __restrict__ attn, int N)
{
  float* row = attn + (size_t)blockIdx.x * N;
  __shared__ float sh[256];
  float m = -3.0e38f;
  for (int j = threadIdx.x; j < N; j += 256) m = fmaxf(m, row[j]);
  sh[threadIdx.x] = m; __syncthreads();
  for (int st = 128; st > 0; st >>= 1) {
    if ((int)threadIdx.x < st)
      sh[threadIdx.x] = fmaxf(sh[threadIdx.x], sh[threadIdx.x + st]);
    __syncthreads();
  }
  m = sh[0]; __syncthreads();
  float s = 0.f;
  for (int j = threadIdx.x; j < N; j += 256) {
    float e = __expf(row[j] - m);
    row[j] = e; s += e;
  }
  sh[threadIdx.x] = s; __syncthreads();
  for (int st = 128; st > 0; st >>= 1) {
    if ((int)threadIdx.x < st) sh[threadIdx.x] += sh[threadIdx.x + st];
    __syncthreads();
  }
  const float inv = 1.0f / sh[0];
  for (int j = threadIdx.x; j < N; j += 256) row[j] *= inv;
}

// Column sums (sum over axis=1) of attn[N,N]; coalesced across threads.
__global__ void __launch_bounds__(256)
colsum_kernel(const float* __restrict__ attn, float* __restrict__ colsum, int N)
{
  const int m = blockIdx.x * 256 + threadIdx.x;
  float s = 0.f;
  for (int n = 0; n < N; ++n) s += attn[(size_t)n * N + m];
  colsum[m] = s;
}

// ---------------------------------------------------------------------------
// Host-side orchestration
// ---------------------------------------------------------------------------
extern "C" void kernel_launch(void* const* d_in, const int* in_sizes, int n_in,
                              void* d_out, int out_size, void* d_ws, size_t ws_size,
                              hipStream_t stream)
{
  (void)in_sizes; (void)n_in; (void)out_size; (void)ws_size;
  const int Bn = BATCH, C = C_OUT, N = NPTS, D = DQK, Ll = NLAYR;

  const float* features = (const float*)d_in[0];
  const float* w1  = (const float*)d_in[1];
  const float* g1  = (const float*)d_in[2];
  const float* b1  = (const float*)d_in[3];
  const float* w2  = (const float*)d_in[4];
  const float* g2  = (const float*)d_in[5];
  const float* b2  = (const float*)d_in[6];
  const float* qkw = (const float*)d_in[7];
  const float* vw  = (const float*)d_in[8];
  const float* vb  = (const float*)d_in[9];
  const float* tw  = (const float*)d_in[10];
  const float* tb  = (const float*)d_in[11];
  const float* bng = (const float*)d_in[12];
  const float* bnb = (const float*)d_in[13];
  float* out = (float*)d_out;

  // ---- carve workspace (256B aligned chunks) ----
  char* base = (char*)d_ws;
  size_t woff = 0;
  auto carve = [&](size_t bytes) -> void* {
    void* p = base + woff;
    woff += (bytes + 255) & ~(size_t)255;
    return p;
  };
  unsigned short* featT_bf = (unsigned short*)carve((size_t)Bn * N * C_IN * 2);
  unsigned short* w1_bf    = (unsigned short*)carve((size_t)C * C_IN * 2);
  unsigned short* w2_bf    = (unsigned short*)carve((size_t)C * C * 2);
  unsigned short* qkw_bf   = (unsigned short*)carve((size_t)Ll * D * C * 2);
  unsigned short* vw_bf    = (unsigned short*)carve((size_t)Ll * C * C * 2);
  unsigned short* tw_bf    = (unsigned short*)carve((size_t)Ll * C * C * 2);
  float*          tmp      = (float*)carve((size_t)Bn * C * N * 4);
  float*          cur      = (float*)carve((size_t)Bn * C * N * 4);
  unsigned short* curT_bf  = (unsigned short*)carve((size_t)Bn * N * C * 2);
  float*          Ktf      = (float*)carve((size_t)Bn * N * D * 4);
  unsigned short* Ktb      = (unsigned short*)carve((size_t)Bn * N * D * 2);
  unsigned short* Vb       = (unsigned short*)carve((size_t)Bn * C * N * 2);
  unsigned short* offT_bf  = (unsigned short*)carve((size_t)Bn * N * C * 2);
  float*          attn     = (float*)carve((size_t)N * N * 4);          // per-batch
  unsigned short* attnbT   = (unsigned short*)carve((size_t)N * N * 2); // per-batch
  float*          colsum   = (float*)carve((size_t)N * 4);
  float*          mean     = (float*)carve(C * 4);
  float*          rstd     = (float*)carve(C * 4);

  auto conv = [&](const float* src, unsigned short* dst, int total) {
    convert_bf16_kernel<<<(total + 255) / 256, 256, 0, stream>>>(src, dst, total);
  };
  auto transp = [&](const float* src, unsigned short* dst, int R, int Cc,
                    long long sIn, long long sOut, const float* scale, int nb) {
    dim3 grid(Cc / 32, R / 32, nb);
    transpose_scale_bf16<<<grid, dim3(32, 8), 0, stream>>>(src, dst, R, Cc, sIn, sOut, scale);
  };
  auto gemm = [&](int mode, const unsigned short* A, const unsigned short* BT,
                  float* Cc, const float* bias, const float* resid,
                  int M, int Nc, int Kd, int lda, int ldbt, int ldc, int ldr,
                  long long sA, long long sBT, long long sC, long long sR, int nb) {
    dim3 grid(Nc / 64, M / 128, nb);
    if (mode == 0)
      gemm_bf16_wmma<0><<<grid, dim3(128), 0, stream>>>(
          A, BT, Cc, bias, resid, M, Nc, Kd, lda, ldbt, ldc, ldr, sA, sBT, sC, sR);
    else if (mode == 1)
      gemm_bf16_wmma<1><<<grid, dim3(128), 0, stream>>>(
          A, BT, Cc, bias, resid, M, Nc, Kd, lda, ldbt, ldc, ldr, sA, sBT, sC, sR);
    else
      gemm_bf16_wmma<2><<<grid, dim3(128), 0, stream>>>(
          A, BT, Cc, bias, resid, M, Nc, Kd, lda, ldbt, ldc, ldr, sA, sBT, sC, sR);
  };
  auto bn = [&](const float* x, const float* g, const float* bb,
                float* of, float* o2, long long o2bs) {
    bn_stats_kernel<<<C, 256, 0, stream>>>(x, mean, rstd, C, N, Bn, 1e-5f);
    const int total = Bn * C * N;
    bn_apply_relu<<<(total + 255) / 256, 256, 0, stream>>>(
        x, mean, rstd, g, bb, of, o2, o2bs, C, N, total);
  };

  // ---- one-time packs: weights (bf16) + transposed features ----
  conv(w1, w1_bf, C * C_IN);
  conv(w2, w2_bf, C * C);
  conv(qkw, qkw_bf, Ll * D * C);
  conv(vw, vw_bf, Ll * C * C);
  conv(tw, tw_bf, Ll * C * C);
  transp(features, featT_bf, C_IN, N,
         (long long)C_IN * N, (long long)N * C_IN, nullptr, Bn);

  // ---- LBR stem: conv1x1 -> BN -> ReLU -> conv1x1 -> BN -> ReLU ----
  gemm(0, w1_bf, featT_bf, tmp, nullptr, nullptr,
       C, N, C_IN, C_IN, C_IN, N, 0,
       0, (long long)N * C_IN, (long long)C * N, 0, Bn);
  bn(tmp, g1, b1, cur, nullptr, 0);
  transp(cur, curT_bf, C, N, (long long)C * N, (long long)N * C, nullptr, Bn);
  gemm(0, w2_bf, curT_bf, tmp, nullptr, nullptr,
       C, N, C, C, C, N, 0,
       0, (long long)N * C, (long long)C * N, 0, Bn);
  bn(tmp, g2, b2, cur, nullptr, 0);
  transp(cur, curT_bf, C, N, (long long)C * N, (long long)N * C, nullptr, Bn);

  // ---- stacked offset attention ----
  for (int i = 0; i < Ll; ++i) {
    // Kt[n,d] = sum_c x[c,n] qk_w[d,c]  ->  A = x^T (curT), B^T = qk_w[i]
    gemm(0, curT_bf, qkw_bf + (size_t)i * D * C, Ktf, nullptr, nullptr,
         N, D, C, C, C, D, 0,
         (long long)N * C, 0, (long long)N * D, 0, Bn);
    conv(Ktf, Ktb, Bn * N * D);

    // V = v_w[i] @ x + v_b[i]  (row-major f32 in tmp, bf16 pack as A-operand)
    gemm(1, vw_bf + (size_t)i * C * C, curT_bf, tmp, vb + (size_t)i * C, nullptr,
         C, N, C, C, C, N, 0,
         0, (long long)N * C, (long long)C * N, 0, Bn);
    conv(tmp, Vb, Bn * C * N);

    // Per-batch attention: 64MB f32 map + 32MB bf16 map stay L2-resident.
    for (int b = 0; b < Bn; ++b) {
      // logits = Kt * Kt^T : self-transpose GEMM (A = B^T = Ktb[b])
      gemm(0, Ktb + (size_t)b * N * D, Ktb + (size_t)b * N * D, attn,
           nullptr, nullptr,
           N, N, D, D, D, N, 0, 0, 0, 0, 0, 1);
      softmax_row_kernel<<<N, 256, 0, stream>>>(attn, N);
      colsum_kernel<<<N / 256, 256, 0, stream>>>(attn, colsum, N);
      // attnbT[m,n] = attn[n,m] / (1e-9 + colsum[m])  (B^T operand, bf16)
      transp(attn, attnbT, N, N, 0, 0, colsum, 1);
      // off = cur - V @ attn   (fused residual epilogue) -> tmp[b]
      gemm(2, Vb + (size_t)b * C * N, attnbT, tmp + (size_t)b * C * N,
           nullptr, cur + (size_t)b * C * N,
           C, N, N, N, N, N, N, 0, 0, 0, 0, 1);
    }
    // off^T in bf16 for the T projection
    transp(tmp, offT_bf, C, N, (long long)C * N, (long long)N * C, nullptr, Bn);

    // t = t_w[i] @ off + t_b[i] -> tmp; BN+ReLU -> cur (+ output slice)
    gemm(1, tw_bf + (size_t)i * C * C, offT_bf, tmp, tb + (size_t)i * C, nullptr,
         C, N, C, C, C, N, 0,
         0, (long long)N * C, (long long)C * N, 0, Bn);
    bn(tmp, bng + (size_t)i * C, bnb + (size_t)i * C,
       cur, out + (size_t)i * C * N, (long long)Ll * C * N);
    transp(cur, curT_bf, C, N, (long long)C * N, (long long)N * C, nullptr, Bn);
  }
}
